// EPARNNModel_85873576116906
// MI455X (gfx1250) — matrix operand order repaired and checked by reading, not compiled, not run on hardware
//
#include <hip/hip_runtime.h>
#include <hip/hip_bf16.h>

typedef __attribute__((ext_vector_type(16))) _Float16 v16h;
typedef __attribute__((ext_vector_type(8)))  _Float16 v8h;
typedef __attribute__((ext_vector_type(8)))  float    v8f;
typedef __attribute__((ext_vector_type(4)))  int      v4i;

#define HID 8
#define SEQ 609
#define IND 5
#define BATCH 2048

#define K1   4872            // SEQ*HID
#define K1P  4896            // padded to 32
#define N1   4096
#define N2   1024
#define N3   609
#define N3P  640             // padded to 128

#define AS1 __attribute__((address_space(1)))
#define AS3 __attribute__((address_space(3)))

#if defined(__has_builtin)
#if __has_builtin(__builtin_amdgcn_global_load_async_to_lds_b128)
#define HAVE_ASYNC_LDS 1
#endif
#endif

// ---- 16B global->LDS copy: async path when available ----
__device__ __forceinline__ void copy16B(const _Float16* g, _Float16* l) {
#ifdef HAVE_ASYNC_LDS
    __builtin_amdgcn_global_load_async_to_lds_b128((AS1 v4i*)g, (AS3 v4i*)l, 0, 0);
#else
    *(uint4*)l = *(const uint4*)g;
#endif
}

template<int N> __device__ __forceinline__ void wait_async() {
#ifdef HAVE_ASYNC_LDS
#if __has_builtin(__builtin_amdgcn_s_wait_asynccnt)
    __builtin_amdgcn_s_wait_asynccnt(N);
#else
    asm volatile("s_wait_asynccnt %0" :: "n"(N) : "memory");
#endif
#endif
}

__device__ __forceinline__ float fsig(float x) {
    return __builtin_amdgcn_rcpf(1.0f + __expf(-x));
}
__device__ __forceinline__ float ftanh_fast(float x) {
    return 2.0f * __builtin_amdgcn_rcpf(1.0f + __expf(-2.0f * x)) - 1.0f;
}

// -------------------- 3-layer fused LSTM --------------------
// 8 lanes per batch element; lane j owns hidden unit j (gate rows j, j+8, j+16, j+24).
// Writes f16 output with row stride K1P (zero-padded tail) for direct WMMA consumption.
__global__ void lstm3_fused(
    const float* __restrict__ x,
    const float* __restrict__ Wih1, const float* __restrict__ Whh1,
    const float* __restrict__ bih1, const float* __restrict__ bhh1,
    const float* __restrict__ Wih2, const float* __restrict__ Whh2,
    const float* __restrict__ bih2, const float* __restrict__ bhh2,
    const float* __restrict__ Wih3, const float* __restrict__ Whh3,
    const float* __restrict__ bih3, const float* __restrict__ bhh3,
    _Float16* __restrict__ out)   // [B, K1P]
{
    const int tid  = threadIdx.x;
    const int j    = tid & 7;
    const int b    = blockIdx.x * 32 + (tid >> 3);
    const int lane = tid & 31;
    const int base = lane & ~7;

    float wi1[4][IND], wh1[4][HID];
    float wi2[4][HID], wh2[4][HID];
    float wi3[4][HID], wh3[4][HID];
    float bs1[4], bs2[4], bs3[4];
#pragma unroll
    for (int g = 0; g < 4; ++g) {
        const int row = g * 8 + j;
#pragma unroll
        for (int k = 0; k < IND; ++k) wi1[g][k] = Wih1[row * IND + k];
#pragma unroll
        for (int k = 0; k < HID; ++k) {
            wh1[g][k] = Whh1[row * HID + k];
            wi2[g][k] = Wih2[row * HID + k];
            wh2[g][k] = Whh2[row * HID + k];
            wi3[g][k] = Wih3[row * HID + k];
            wh3[g][k] = Whh3[row * HID + k];
        }
        bs1[g] = bih1[row] + bhh1[row];
        bs2[g] = bih2[row] + bhh2[row];
        bs3[g] = bih3[row] + bhh3[row];
    }

    float h1v[HID], h2v[HID], h3v[HID];
#pragma unroll
    for (int k = 0; k < HID; ++k) { h1v[k] = 0.f; h2v[k] = 0.f; h3v[k] = 0.f; }
    float c1 = 0.f, c2 = 0.f, c3 = 0.f;

    const float*   xb = x   + (size_t)b * SEQ * IND;
    _Float16*      ob = out + (size_t)b * K1P;

    // zero the K-pad tail (cols K1..K1P-1): 24 halves, 3 per lane
#pragma unroll
    for (int p = 0; p < 3; ++p) ob[K1 + j * 3 + p] = (_Float16)0.0f;

    for (int t = 0; t < SEQ; ++t) {
        float xv[IND];
#pragma unroll
        for (int k = 0; k < IND; ++k) xv[k] = xb[t * IND + k];

        // ---- layer 1 ----
        float g1[4];
#pragma unroll
        for (int g = 0; g < 4; ++g) {
            float a = bs1[g];
#pragma unroll
            for (int k = 0; k < IND; ++k) a = fmaf(wi1[g][k], xv[k], a);
#pragma unroll
            for (int k = 0; k < HID; ++k) a = fmaf(wh1[g][k], h1v[k], a);
            g1[g] = a;
        }
        {
            float i = fsig(g1[0]), f = fsig(g1[1]);
            float g = ftanh_fast(g1[2]), o = fsig(g1[3]);
            c1 = f * c1 + i * g;
            float h = o * ftanh_fast(c1);
#pragma unroll
            for (int k = 0; k < HID; ++k) h1v[k] = __shfl(h, base + k, 32);
        }
        // ---- layer 2 ----
        float g2[4];
#pragma unroll
        for (int g = 0; g < 4; ++g) {
            float a = bs2[g];
#pragma unroll
            for (int k = 0; k < HID; ++k) a = fmaf(wi2[g][k], h1v[k], a);
#pragma unroll
            for (int k = 0; k < HID; ++k) a = fmaf(wh2[g][k], h2v[k], a);
            g2[g] = a;
        }
        {
            float i = fsig(g2[0]), f = fsig(g2[1]);
            float g = ftanh_fast(g2[2]), o = fsig(g2[3]);
            c2 = f * c2 + i * g;
            float h = o * ftanh_fast(c2);
#pragma unroll
            for (int k = 0; k < HID; ++k) h2v[k] = __shfl(h, base + k, 32);
        }
        // ---- layer 3 ----
        float g3[4];
#pragma unroll
        for (int g = 0; g < 4; ++g) {
            float a = bs3[g];
#pragma unroll
            for (int k = 0; k < HID; ++k) a = fmaf(wi3[g][k], h2v[k], a);
#pragma unroll
            for (int k = 0; k < HID; ++k) a = fmaf(wh3[g][k], h3v[k], a);
            g3[g] = a;
        }
        {
            float i = fsig(g3[0]), f = fsig(g3[1]);
            float g = ftanh_fast(g3[2]), o = fsig(g3[3]);
            c3 = f * c3 + i * g;
            float h = o * ftanh_fast(c3);
#pragma unroll
            for (int k = 0; k < HID; ++k) h3v[k] = __shfl(h, base + k, 32);
            ob[t * HID + j] = (_Float16)h;
        }
    }
}

// -------------------- f32 -> f16 weight conversion with zero padding --------------------
__global__ __launch_bounds__(256) void cvt_pad_f16(
    const float* __restrict__ src, _Float16* __restrict__ dst,
    int N, int K, int Np, int Kp)
{
    int idx = blockIdx.x * 256 + threadIdx.x;
    if (idx >= Np * Kp) return;
    int n = idx / Kp, k = idx - n * Kp;
    float v = (n < N && k < K) ? src[(size_t)n * K + k] : 0.0f;
    dst[idx] = (_Float16)v;
}

// -------------------- WMMA GEMM: C = act(A * B^T + bias) --------------------
// A: [M, lda] f16 (padded), B: [Npad, ldb] f16 (padded). Block tile 128x128,
// 8 waves, each wave 32(M) x 64(N) = 8 wmma per 32-K chunk. Double-buffered LDS
// fed by GLOBAL_LOAD_ASYNC_TO_LDS_B128 when available.
template<bool RELU, typename OutT>
__global__ __launch_bounds__(256) void gemm_bias_wmma(
    const _Float16* __restrict__ A, int lda,
    const _Float16* __restrict__ B, int ldb,
    const float* __restrict__ bias,
    OutT* __restrict__ C, int N, int Kp)
{
    constexpr int LDH = 40;   // 32 halves + 8 pad (80B rows: conflict-free frag reads)
    __shared__ __align__(16) _Float16 As[2][128 * LDH];
    __shared__ __align__(16) _Float16 Bs[2][128 * LDH];

    const int tid  = threadIdx.x;
    const int lane = tid & 31;
    const int wid  = tid >> 5;
    const int wm   = wid & 3;          // 4 waves along M (32 each)
    const int wn   = wid >> 2;         // 2 waves along N (64 each)
    const int m0   = blockIdx.y * 128;
    const int n0   = blockIdx.x * 128;

    // staging: 128 rows x 4 segs of 8 halves = 512 chunks per tile; 2 per thread
    const int srow = tid >> 2;              // 0..63 (second chunk: +64)
    const int sseg = (tid & 3) * 8;

    v8f acc[2][4] = {};

    auto stage = [&](int buf, int k0) {
        const _Float16* gA0 = A + (size_t)(m0 + srow)      * lda + k0 + sseg;
        const _Float16* gA1 = A + (size_t)(m0 + srow + 64) * lda + k0 + sseg;
        const _Float16* gB0 = B + (size_t)(n0 + srow)      * ldb + k0 + sseg;
        const _Float16* gB1 = B + (size_t)(n0 + srow + 64) * ldb + k0 + sseg;
        copy16B(gA0, &As[buf][srow * LDH + sseg]);
        copy16B(gA1, &As[buf][(srow + 64) * LDH + sseg]);
        copy16B(gB0, &Bs[buf][srow * LDH + sseg]);
        copy16B(gB1, &Bs[buf][(srow + 64) * LDH + sseg]);
    };

    stage(0, 0);
    const int nk = Kp >> 5;
    for (int ki = 0; ki < nk; ++ki) {
        const int buf = ki & 1;
        if (ki + 1 < nk) {
            stage(buf ^ 1, (ki + 1) << 5);
            wait_async<4>();    // previous tile's 4 copies complete; next 4 in flight
        } else {
            wait_async<0>();
        }
        __syncthreads();

        // A frag: lanes 0-15 hold K{0..7,16..23}, lanes 16-31 K{8..15,24..31}; lane&15 = M
        const int koffA = (lane & 16) ? 8 : 0;
        const int mrow  = wm * 32 + (lane & 15);
        v16h afrag[2];
#pragma unroll
        for (int i = 0; i < 2; ++i) {
            const _Float16* p = &As[buf][(mrow + i * 16) * LDH];
            v8h lo = *(const v8h*)(p + koffA);
            v8h hi = *(const v8h*)(p + 16 + koffA);
#pragma unroll
            for (int q = 0; q < 8; ++q) { afrag[i][q] = lo[q]; afrag[i][q + 8] = hi[q]; }
        }
        // B frag: lanes 0-15 hold K{0..7,16..23}, lanes 16-31 K{8..15,24..31}; lane&15 = N
        const int koffB = (lane & 16) ? 8 : 0;
        const int nrow  = wn * 64 + (lane & 15);
        v16h bfrag[4];
#pragma unroll
        for (int i = 0; i < 4; ++i) {
            const _Float16* p = &Bs[buf][(nrow + i * 16) * LDH + koffB];
            v8h lo = *(const v8h*)p;
            v8h hi = *(const v8h*)(p + 16);
#pragma unroll
            for (int q = 0; q < 8; ++q) { bfrag[i][q] = lo[q]; bfrag[i][q + 8] = hi[q]; }
        }

#pragma unroll
        for (int i = 0; i < 2; ++i)
#pragma unroll
            for (int jn = 0; jn < 4; ++jn)
                acc[i][jn] = __builtin_amdgcn_wmma_f32_16x16x32_f16(
                    false, afrag[i], false, bfrag[jn],
                    (short)0, acc[i][jn], false, false);
        __syncthreads();
    }

    // epilogue: bias (+ReLU); C layout: VGPR r -> M=r (lanes<16) / r+8, lane&15 = N
#pragma unroll
    for (int i = 0; i < 2; ++i) {
#pragma unroll
        for (int jn = 0; jn < 4; ++jn) {
            const int n = n0 + wn * 64 + jn * 16 + (lane & 15);
            if (n < N) {
                const float bv = bias[n];
                const int mbase = m0 + wm * 32 + i * 16 + ((lane & 16) ? 8 : 0);
#pragma unroll
                for (int r = 0; r < 8; ++r) {
                    float v = acc[i][jn][r] + bv;
                    if (RELU) v = fmaxf(v, 0.0f);
                    C[(size_t)(mbase + r) * N + n] = (OutT)v;
                }
            }
        }
    }
}

// -------------------- launch --------------------
extern "C" void kernel_launch(void* const* d_in, const int* in_sizes, int n_in,
                              void* d_out, int out_size, void* d_ws, size_t ws_size,
                              hipStream_t stream)
{
    (void)in_sizes; (void)n_in; (void)out_size; (void)ws_size;
    const float* x    = (const float*)d_in[0];
    const float* Wih1 = (const float*)d_in[1];
    const float* Whh1 = (const float*)d_in[2];
    const float* bih1 = (const float*)d_in[3];
    const float* bhh1 = (const float*)d_in[4];
    const float* Wih2 = (const float*)d_in[5];
    const float* Whh2 = (const float*)d_in[6];
    const float* bih2 = (const float*)d_in[7];
    const float* bhh2 = (const float*)d_in[8];
    const float* Wih3 = (const float*)d_in[9];
    const float* Whh3 = (const float*)d_in[10];
    const float* bih3 = (const float*)d_in[11];
    const float* bhh3 = (const float*)d_in[12];
    const float* W1   = (const float*)d_in[13];
    const float* b1   = (const float*)d_in[14];
    const float* W2   = (const float*)d_in[15];
    const float* b2   = (const float*)d_in[16];
    const float* W3   = (const float*)d_in[17];
    const float* b3   = (const float*)d_in[18];

    char* ws = (char*)d_ws;
    auto align256 = [](size_t v) { return (v + 255) & ~(size_t)255; };
    size_t off = 0;
    _Float16* lstmh = (_Float16*)(ws + off); off += align256((size_t)BATCH * K1P * sizeof(_Float16));
    _Float16* W1h   = (_Float16*)(ws + off); off += align256((size_t)N1 * K1P * sizeof(_Float16));
    _Float16* W2h   = (_Float16*)(ws + off); off += align256((size_t)N2 * N1 * sizeof(_Float16));
    _Float16* W3h   = (_Float16*)(ws + off); off += align256((size_t)N3P * N2 * sizeof(_Float16));
    _Float16* h1h   = (_Float16*)(ws + off); off += align256((size_t)BATCH * N1 * sizeof(_Float16));
    _Float16* h2h   = (_Float16*)(ws + off); off += align256((size_t)BATCH * N2 * sizeof(_Float16));

    // weight conversions (f32 -> f16, zero-padded)
    {
        int e1 = N1 * K1P;  cvt_pad_f16<<<(e1 + 255) / 256, 256, 0, stream>>>(W1, W1h, N1, K1, N1, K1P);
        int e2 = N2 * N1;   cvt_pad_f16<<<(e2 + 255) / 256, 256, 0, stream>>>(W2, W2h, N2, N1, N2, N1);
        int e3 = N3P * N2;  cvt_pad_f16<<<(e3 + 255) / 256, 256, 0, stream>>>(W3, W3h, N3, N2, N3P, N2);
    }

    // LSTM: 32 batches per 256-thread block -> padded f16 activations
    lstm3_fused<<<BATCH / 32, 256, 0, stream>>>(
        x, Wih1, Whh1, bih1, bhh1, Wih2, Whh2, bih2, bhh2,
        Wih3, Whh3, bih3, bhh3, lstmh);

    // FC1: [2048 x 4896] * [4896 x 4096] + b1, ReLU -> f16
    gemm_bias_wmma<true, _Float16><<<dim3(N1 / 128, BATCH / 128), 256, 0, stream>>>(
        lstmh, K1P, W1h, K1P, b1, h1h, N1, K1P);
    // FC2: [2048 x 4096] * [4096 x 1024] + b2, ReLU -> f16
    gemm_bias_wmma<true, _Float16><<<dim3(N2 / 128, BATCH / 128), 256, 0, stream>>>(
        h1h, N1, W2h, N1, b2, h2h, N2, N1);
    // FC3: [2048 x 1024] * [1024 x 609] + b3 -> f32 d_out
    gemm_bias_wmma<false, float><<<dim3(N3P / 128, BATCH / 128), 256, 0, stream>>>(
        h2h, N2, W3h, N2, b3, (float*)d_out, N3, N2);
}
